// GNNClassifier_56418690400931
// MI455X (gfx1250) — matrix operand order repaired and checked by reading, not compiled, not run on hardware
//
#include <hip/hip_runtime.h>
#include <hip/hip_bf16.h>

typedef __attribute__((ext_vector_type(16))) __bf16 v16bf;
typedef __attribute__((ext_vector_type(8)))  float  v8f;

#define HDIM 128
#define BN_EPS 1e-5f

// ---------- fp32x2 -> packed bf16, RNE (used in cold packing/epilogue kernels) ----------
__device__ __forceinline__ unsigned pk_bf16_rne(float a, float b) {
  unsigned ua = __builtin_bit_cast(unsigned, a);
  unsigned ub = __builtin_bit_cast(unsigned, b);
  ua = (ua + 0x7FFFu + ((ua >> 16) & 1u)) >> 16;     // round-to-nearest-even
  ub = (ub + 0x7FFFu + ((ub >> 16) & 1u)) >> 16;
  return (ub << 16) | (ua & 0xFFFFu);
}

// ---------- fp32x2 -> packed bf16, truncating: single v_perm_b32 (hot GEMM path) ----------
// result bytes: [1:0] = a[3:2], [3:2] = b[3:2]  -> low half = bf16(a), high = bf16(b)
__device__ __forceinline__ unsigned pk_bf16_fast(float a, float b) {
  return __builtin_amdgcn_perm(__builtin_bit_cast(unsigned, b),
                               __builtin_bit_cast(unsigned, a), 0x07060302u);
}

// ---------- weight packing: W (K x Ncols, fp32 row-major) -> bf16 in WMMA B layout ----------
// dword t: j = t&7 (VGPR), lane = (t>>3)&31, nt = (t>>8)%NT, kt = (t>>8)/NT
// B 32x16 tile: lane half h, VGPR j<4 -> K=8h+2j,+1; VGPR j>=4 -> K=16+8h+2(j-4),+1
__global__ void pack_B_kernel(const float* __restrict__ W, unsigned* __restrict__ out,
                              int Ncols, int NT, int total) {
  int t = blockIdx.x * blockDim.x + threadIdx.x;
  if (t >= total) return;
  int j    = t & 7;
  int lane = (t >> 3) & 31;
  int rest = t >> 8;
  int nt   = rest % NT;
  int kt   = rest / NT;
  int k = kt * 32 + ((j >> 2) << 4) + ((lane >> 4) << 3) + 2 * (j & 3);
  int n = nt * 16 + (lane & 15);
  out[t] = pk_bf16_rne(W[(size_t)k * Ncols + n], W[(size_t)(k + 1) * Ncols + n]);
}

// ---------- degree ----------
__global__ void deg_init_kernel(float* __restrict__ deg, int N) {
  int i = blockIdx.x * blockDim.x + threadIdx.x;
  if (i < N) deg[i] = 1.0f;   // self-loop
}
__global__ void deg_edges_kernel(const int* __restrict__ ei, float* __restrict__ deg, int E) {
  int e = blockIdx.x * blockDim.x + threadIdx.x;
  if (e < E) atomicAdd(&deg[ei[E + e]], 1.0f);   // dst row
}
__global__ void deg_rsqrt_kernel(float* __restrict__ deg, int N) {
  int i = blockIdx.x * blockDim.x + threadIdx.x;
  if (i < N) deg[i] = rsqrtf(deg[i]);
}

// ---------- WMMA GEMM: out[M x NT*16] = A[M x K] * Bpacked ----------
// One wave computes MT row-tiles x NT col-tiles. B regs loaded once per K-step,
// reused across all MT row tiles. A is fp32 (converted via v_perm) or packed bf16.
template <int MT, int NT, bool ABF16>
__global__ __launch_bounds__(64)
void gemm_wmma_kernel(const void* __restrict__ Araw, const unsigned* __restrict__ Bp,
                      float* __restrict__ out, const float* __restrict__ bias,
                      int M, int K, int relu) {
  const int wave  = threadIdx.x >> 5;
  const int lane  = threadIdx.x & 31;
  const int group = blockIdx.x * 2 + wave;            // wave-uniform
  const int tile0 = group * MT;                       // first 16-row tile
  const int Mtiles = M >> 4;
  if (tile0 >= Mtiles) return;                        // uniform: EXEC stays all-ones

  const int koff8 = (lane >> 4) * 8;                  // A layout: lanes>=16 -> K+8 in each half
  size_t rowoff[MT];
  bool valid[MT];
#pragma unroll
  for (int mt = 0; mt < MT; ++mt) {
    valid[mt] = (tile0 + mt) < Mtiles;                // wave-uniform
    int r = (tile0 + mt) * 16 + (lane & 15);
    rowoff[mt] = (size_t)(valid[mt] ? r : 0) * K;
  }

  v8f acc[MT][NT] = {};

  for (int kb = 0; kb < K; kb += 32) {
    // ---- load B tiles for this K-step (shared across all MT row tiles) ----
    union { uint4 q[2]; v16bf v; } b[NT];
#pragma unroll
    for (int nt = 0; nt < NT; ++nt) {
      const uint4* bp = (const uint4*)Bp + ((size_t)(((kb >> 5) * NT + nt) * 32 + lane)) * 2;
      b[nt].q[0] = bp[0];
      b[nt].q[1] = bp[1];
    }
    // ---- per row tile: load (and maybe convert) A, run NT WMMAs ----
#pragma unroll
    for (int mt = 0; mt < MT; ++mt) {
      if (valid[mt]) {
        union { unsigned u[8]; uint4 q[2]; v16bf v; } a;
        if (ABF16) {
          // packed bf16 A: lane's halves [koff8 .. +7] and [16+koff8 .. +7]
          const uint4* ap = (const uint4*)((const unsigned short*)Araw + rowoff[mt] + kb + koff8);
          a.q[0] = ap[0];                             // K: koff8+0..7   -> VGPR 0..3
          a.q[1] = ap[2];                             // K: 16+koff8+0..7-> VGPR 4..7
        } else {
          const float* ap = (const float*)Araw + rowoff[mt] + kb + koff8;
          float4 f0 = *(const float4*)(ap + 0);       // K: koff8+0..3   -> VGPR 0,1
          float4 f1 = *(const float4*)(ap + 4);       // K: koff8+4..7   -> VGPR 2,3
          float4 f2 = *(const float4*)(ap + 16);      // K: 16+koff8+0..3-> VGPR 4,5
          float4 f3 = *(const float4*)(ap + 20);      // K: 16+koff8+4..7-> VGPR 6,7
          a.u[0] = pk_bf16_fast(f0.x, f0.y); a.u[1] = pk_bf16_fast(f0.z, f0.w);
          a.u[2] = pk_bf16_fast(f1.x, f1.y); a.u[3] = pk_bf16_fast(f1.z, f1.w);
          a.u[4] = pk_bf16_fast(f2.x, f2.y); a.u[5] = pk_bf16_fast(f2.z, f2.w);
          a.u[6] = pk_bf16_fast(f3.x, f3.y); a.u[7] = pk_bf16_fast(f3.z, f3.w);
        }
#pragma unroll
        for (int nt = 0; nt < NT; ++nt) {
          acc[mt][nt] = __builtin_amdgcn_wmma_f32_16x16x32_bf16(
              false, a.v, false, b[nt].v, (short)0, acc[mt][nt], false, false);
        }
      }
    }
  }

  // ---- epilogue: C layout -> row-major fp32 store ----
  const int mrow  = (lane >> 4) * 8;                  // lanes>=16 -> M+8
  const int col   = lane & 15;
  const int ncols = NT * 16;
#pragma unroll
  for (int mt = 0; mt < MT; ++mt) {
    if (valid[mt]) {
      int rowbase = (tile0 + mt) * 16;
#pragma unroll
      for (int nt = 0; nt < NT; ++nt) {
        int c = nt * 16 + col;
        float badd = bias ? bias[c] : 0.0f;
#pragma unroll
        for (int v = 0; v < 8; ++v) {
          float val = acc[mt][nt][v] + badd;
          if (relu) val = fmaxf(val, 0.0f);
          out[(size_t)(rowbase + mrow + v) * ncols + c] = val;
        }
      }
    }
  }
}

// ---------- self-loop term: agg = (1/deg) * lin ----------
__global__ void self_init_kernel(const float* __restrict__ lin, const float* __restrict__ dis,
                                 float* __restrict__ agg, int total) {
  int idx = blockIdx.x * blockDim.x + threadIdx.x;
  if (idx >= total) return;
  int n = idx >> 7;
  float di = dis[n];
  agg[idx] = di * di * lin[idx];
}

// ---------- edge scatter: one wave per edge, lane covers 4 features ----------
__global__ __launch_bounds__(256)
void edge_scatter_kernel(const int* __restrict__ ei, const float* __restrict__ dis,
                         const float* __restrict__ lin, float* __restrict__ agg, int E) {
  int e = blockIdx.x * 8 + (threadIdx.x >> 5);
  if (e >= E) return;
  int lane = threadIdx.x & 31;
  int s = ei[e];
  int d = ei[E + e];
  float nrm = dis[s] * dis[d];
  float4 h = ((const float4*)(lin + (size_t)s * HDIM))[lane];
  float* dst = agg + (size_t)d * HDIM + lane * 4;
  atomicAdd(dst + 0, h.x * nrm);
  atomicAdd(dst + 1, h.y * nrm);
  atomicAdd(dst + 2, h.z * nrm);
  atomicAdd(dst + 3, h.w * nrm);
}

// ---------- bias + BN(eval) + ReLU -> packed bf16 activations ----------
// thread idx handles features (2*idx)&127, +1 of node idx>>6; writes one dword.
__global__ void bn_relu_bf16_kernel(const float* __restrict__ agg, const float* __restrict__ b,
                                    const float* __restrict__ gamma, const float* __restrict__ beta,
                                    const float* __restrict__ rmean, const float* __restrict__ rvar,
                                    unsigned* __restrict__ act, int total2) {
  int idx = blockIdx.x * blockDim.x + threadIdx.x;
  if (idx >= total2) return;
  int f = (idx << 1) & (HDIM - 1);
  const float* ag = agg + ((size_t)idx << 1);
  float v0 = ag[0] + b[f];
  float v1 = ag[1] + b[f + 1];
  v0 = (v0 - rmean[f])     * rsqrtf(rvar[f]     + BN_EPS) * gamma[f]     + beta[f];
  v1 = (v1 - rmean[f + 1]) * rsqrtf(rvar[f + 1] + BN_EPS) * gamma[f + 1] + beta[f + 1];
  v0 = fmaxf(v0, 0.0f);
  v1 = fmaxf(v1, 0.0f);
  act[idx] = pk_bf16_rne(v0, v1);
}

// ---------- final: out[n] = dot(hc[n][0:64], Wc2) + bc2 ----------
__global__ void final_mlp_kernel(const float* __restrict__ hc, const float* __restrict__ Wc2,
                                 const float* __restrict__ bc2, float* __restrict__ out, int N) {
  int n = blockIdx.x * blockDim.x + threadIdx.x;
  if (n >= N) return;
  const float* h = hc + (size_t)n * 64;
  float s = bc2[0];
#pragma unroll
  for (int k = 0; k < 64; ++k) s += h[k] * Wc2[k];
  out[n] = s;
}

extern "C" void kernel_launch(void* const* d_in, const int* in_sizes, int n_in,
                              void* d_out, int out_size, void* d_ws, size_t ws_size,
                              hipStream_t stream) {
  const float* x   = (const float*)d_in[0];
  const int*   ei  = (const int*)d_in[1];
  const float* W1  = (const float*)d_in[2];
  const float* b1  = (const float*)d_in[3];
  const float* W2  = (const float*)d_in[4];
  const float* b2  = (const float*)d_in[5];
  const float* W3  = (const float*)d_in[6];
  const float* b3  = (const float*)d_in[7];
  const float* gammas = (const float*)d_in[8];
  const float* betas  = (const float*)d_in[9];
  const float* rmeans = (const float*)d_in[10];
  const float* rvars  = (const float*)d_in[11];
  const float* Wc1 = (const float*)d_in[12];
  const float* bc1 = (const float*)d_in[13];
  const float* Wc2 = (const float*)d_in[14];
  const float* bc2 = (const float*)d_in[15];
  float* out = (float*)d_out;

  const int D = 1024;
  const int N = in_sizes[0] / D;       // 100000
  const int E = in_sizes[1] / 2;       // 1600000

  // workspace layout (fixed roles)
  char* base = (char*)d_ws;
  size_t off = 0;
  float*    lin  = (float*)(base + off);    off += (size_t)N * HDIM * 4;  // 51.2 MB fp32
  float*    agg  = (float*)(base + off);    off += (size_t)N * HDIM * 4;  // 51.2 MB fp32
  unsigned* actb = (unsigned*)(base + off); off += (size_t)N * HDIM * 2;  // 25.6 MB bf16
  float*    dis  = (float*)(base + off);    off += (size_t)N * 4;
  off = (off + 255) & ~(size_t)255;
  unsigned* W1p  = (unsigned*)(base + off); off += (size_t)D * HDIM * 2;  // bf16 packed
  unsigned* W2p  = (unsigned*)(base + off); off += (size_t)HDIM * HDIM * 2;
  unsigned* W3p  = (unsigned*)(base + off); off += (size_t)HDIM * HDIM * 2;
  unsigned* Wc1p = (unsigned*)(base + off); off += (size_t)HDIM * 64 * 2;

  // ---- pack weights into WMMA B layout (bf16) ----
  {
    int t1 = D * HDIM / 2;      // 65536 dwords
    pack_B_kernel<<<(t1 + 255) / 256, 256, 0, stream>>>(W1, W1p, HDIM, 8, t1);
    int t2 = HDIM * HDIM / 2;   // 8192
    pack_B_kernel<<<(t2 + 255) / 256, 256, 0, stream>>>(W2, W2p, HDIM, 8, t2);
    pack_B_kernel<<<(t2 + 255) / 256, 256, 0, stream>>>(W3, W3p, HDIM, 8, t2);
    int t3 = HDIM * 64 / 2;     // 4096
    pack_B_kernel<<<(t3 + 255) / 256, 256, 0, stream>>>(Wc1, Wc1p, 64, 4, t3);
  }

  // ---- degree / normalization ----
  deg_init_kernel<<<(N + 255) / 256, 256, 0, stream>>>(dis, N);
  deg_edges_kernel<<<(E + 255) / 256, 256, 0, stream>>>(ei, dis, E);
  deg_rsqrt_kernel<<<(N + 255) / 256, 256, 0, stream>>>(dis, N);

  const int Mtiles   = N / 16;                            // 6250
  const int groups   = (Mtiles + 3) / 4;                  // MT=4 -> 1563 wave-groups
  const int gemmBlks = (groups + 1) / 2;                  // 2 waves per block
  const int totNH    = N * HDIM;
  const int elemBlks = (totNH + 255) / 256;
  const int bnBlks   = (totNH / 2 + 255) / 256;
  const int edgeBlks = (E + 7) / 8;

  const unsigned* Wp[3] = {W1p, W2p, W3p};
  const float* bs[3] = {b1, b2, b3};

  for (int i = 0; i < 3; ++i) {
    if (i == 0) {
      gemm_wmma_kernel<4, 8, false><<<gemmBlks, 64, 0, stream>>>(x, Wp[0], lin, nullptr, N, D, 0);
    } else {
      gemm_wmma_kernel<4, 8, true><<<gemmBlks, 64, 0, stream>>>(actb, Wp[i], lin, nullptr, N, HDIM, 0);
    }
    self_init_kernel<<<elemBlks, 256, 0, stream>>>(lin, dis, agg, totNH);
    edge_scatter_kernel<<<edgeBlks, 256, 0, stream>>>(ei, dis, lin, agg, E);
    bn_relu_bf16_kernel<<<bnBlks, 256, 0, stream>>>(agg, bs[i],
        gammas + i * HDIM, betas + i * HDIM, rmeans + i * HDIM, rvars + i * HDIM,
        actb, totNH / 2);
  }

  // ---- classifier: hc = relu(act3 @ Wc1 + bc1) via WMMA, then dot with Wc2 ----
  float* hc = lin;   // N x 64 fp32, reuse lin buffer
  gemm_wmma_kernel<4, 4, true><<<gemmBlks, 64, 0, stream>>>(actb, Wc1p, hc, bc1, N, HDIM, 1);
  final_mlp_kernel<<<(N + 255) / 256, 256, 0, stream>>>(hc, Wc2, bc2, out, N);

  (void)n_in; (void)ws_size; (void)out_size;
}
